// QBNNBrainTorch_16870631539235
// MI455X (gfx1250) — matrix operand order repaired and checked by reading, not compiled, not run on hardware
//
#include <hip/hip_runtime.h>

// ---------------- types ----------------
typedef __bf16 bf16;
typedef __attribute__((ext_vector_type(16))) __bf16 bf16x16;
typedef __attribute__((ext_vector_type(8)))  float  f32x8;
typedef __attribute__((ext_vector_type(4)))  unsigned int uint32x4;
typedef __attribute__((ext_vector_type(8)))  int int32x8;
typedef __attribute__((ext_vector_type(4)))  int int32x4;

#define NN    2048   // num_neurons
#define BATCH 4096
#define KIN   1024   // max_input_size
#define NOUT  512    // max_output_size
#define TSTEP 8

#define BM 128
#define BN 64
#define BK 32
#define LDA 40       // padded LDS row stride (bf16): 64B row + 16B pad
#define LDB 40

#if defined(__has_builtin)
#  if __has_builtin(__builtin_amdgcn_tensor_load_to_lds)
#    define HAVE_TDM 1
#  else
#    define HAVE_TDM 0
#  endif
#else
#  define HAVE_TDM 0
#endif

#if __has_include(<hip/amd_detail/amd_gfx1250_TDM.h>)
#  define TDM_SIX_ARGS 1
#endif

// ---------------- WMMA fragment loads (wave32 layouts, CDNA5 ISA) ----------------
// A 16x32 bf16: lane L -> M=L&15, half=L>>4; VGPR0..3 = K[half*8..+7], VGPR4..7 = K[16+half*8..+7]
__device__ __forceinline__ bf16x16 load_frag_a(const bf16* lds, int mBase, int lane) {
  const bf16* p = lds + (mBase + (lane & 15)) * LDA + ((lane >> 4) << 3);
  union { bf16x16 v; float4 f4[2]; } f;
  f.f4[0] = *(const float4*)(p);
  f.f4[1] = *(const float4*)(p + 16);
  return f.v;
}
// B 32x16 bf16 stored n-major in LDS: lane L -> N=L&15, half=L>>4; VGPR0..3 = K[half*8..+7], VGPR4..7 = K[16+half*8..+7]
__device__ __forceinline__ bf16x16 load_frag_b(const bf16* lds, int nBase, int lane) {
  const bf16* p = lds + (nBase + (lane & 15)) * LDB + ((lane >> 4) << 3);
  union { bf16x16 v; float4 f4[2]; } f;
  f.f4[0] = *(const float4*)(p);
  f.f4[1] = *(const float4*)(p + 16);
  return f.v;
}

// ---------------- LDS staging (256 threads) ----------------
// bf16 A tile [BM x BK] from row-major global (stride ldg), straight b128 copies
__device__ __forceinline__ void stage_a_bf16(bf16* dst, const bf16* src, int row0, int kt,
                                             int ldg, int tid) {
#pragma unroll
  for (int i = 0; i < 2; ++i) {
    int slot = i * 256 + tid;          // 512 slots x 8 bf16
    int r  = slot >> 2;
    int ko = (slot & 3) << 3;
    float4 d = *(const float4*)(src + (size_t)(row0 + r) * ldg + kt + ko);
    *(float4*)(dst + r * LDA + ko) = d;
  }
}
// f32 A tile [BM x BK] -> bf16 LDS
__device__ __forceinline__ void stage_a_f32(bf16* dst, const float* src, int row0, int kt,
                                            int ldg, int tid) {
#pragma unroll
  for (int i = 0; i < 4; ++i) {
    int slot = i * 256 + tid;          // 1024 slots x 4 f32
    int r  = slot >> 3;
    int ko = (slot & 7) << 2;
    float4 d = *(const float4*)(src + (size_t)(row0 + r) * ldg + kt + ko);
    bf16* o = dst + r * LDA + ko;
    o[0] = (bf16)d.x; o[1] = (bf16)d.y; o[2] = (bf16)d.z; o[3] = (bf16)d.w;
  }
}
// bf16 B tile [BN x BK] from PRE-TRANSPOSED global W^T[n][k] (stride ldg): b128 copies
__device__ __forceinline__ void stage_bt(bf16* dst, const bf16* srcT, int n0, int kt,
                                         int ldg, int tid) {
  int r  = tid >> 2;                   // 256 slots x 8 bf16 (BN*BK/8)
  int ko = (tid & 3) << 3;
  float4 d = *(const float4*)(srcT + (size_t)(n0 + r) * ldg + kt + ko);
  *(float4*)(dst + r * LDB + ko) = d;
}

#define WMMA_BF16(A, B, C) \
  __builtin_amdgcn_wmma_f32_16x16x32_bf16(false, (A), false, (B), (short)0, (C), false, false)

// ---------------- TDM: 2D bf16 tile -> padded LDS (CDNA5 ISA) ----------------
#if HAVE_TDM
__device__ __forceinline__ void tdm_load_tile_2d(unsigned lds_off, const bf16* gtile,
                                                 unsigned tile_k, unsigned tile_n,
                                                 unsigned tensor_k, unsigned tensor_n,
                                                 unsigned row_stride_elems) {
  unsigned long long ga = (unsigned long long)(uintptr_t)gtile;
  uint32x4 g0;
  g0[0] = 1u;                                   // D# count=1, user mode
  g0[1] = lds_off;                              // LDS byte address
  g0[2] = (unsigned)(ga & 0xFFFFFFFFu);         // global_addr[31:0]
  g0[3] = (unsigned)((ga >> 32) & 0x01FFFFFFu)  // global_addr[56:32]
        | (2u << 30);                           // type=2 (image)
  int32x8 g1;
  g1[0] = (int)((1u << 16)                      // data_size: 2 bytes
              | (1u << 20)                      // pad_enable
              | (3u << 22)                      // pad_interval: 16 DW (64B)
              | (3u << 25));                    // pad_amount:   4 DW (16B)
  g1[1] = (int)((tensor_k & 0xFFFFu) << 16);    // tensor_dim0[15:0]
  g1[2] = (int)(((tensor_k >> 16) & 0xFFFFu) | ((tensor_n & 0xFFFFu) << 16));
  g1[3] = (int)(((tensor_n >> 16) & 0xFFFFu) | ((tile_k & 0xFFFFu) << 16)); // tile_dim0
  g1[4] = (int)(tile_n & 0xFFFFu);              // tile_dim1 (tile_dim2=0)
  g1[5] = (int)row_stride_elems;                // tensor_dim0_stride[31:0]
  g1[6] = 0;
  g1[7] = 0;
  int32x4 gz = {0, 0, 0, 0};
#if defined(TDM_SIX_ARGS)
  int32x8 gz8 = {0, 0, 0, 0, 0, 0, 0, 0};
  __builtin_amdgcn_tensor_load_to_lds(g0, g1, gz, gz, gz8, 0);
#else
  __builtin_amdgcn_tensor_load_to_lds(g0, g1, gz, gz, 0);
#endif
}
#endif

// ---------------- prep: [rows x cols] f32 (optionally * mask) -> transposed bf16 [cols x rows] ----
__global__ __launch_bounds__(256)
void transpose_to_bf16_kernel(const float* __restrict__ a, const float* __restrict__ m,
                              bf16* __restrict__ ot, int rows, int cols) {
  __shared__ float t[32][33];
  int bx = blockIdx.x * 32;            // col base (input)
  int by = blockIdx.y * 32;            // row base (input)
  int tx = threadIdx.x & 31, ty = threadIdx.x >> 5;
#pragma unroll
  for (int i = 0; i < 4; ++i) {
    int r = by + ty + i * 8, c = bx + tx;
    float v = a[(size_t)r * cols + c];
    if (m) v *= m[(size_t)r * cols + c];
    t[ty + i * 8][tx] = v;
  }
  __syncthreads();
#pragma unroll
  for (int i = 0; i < 4; ++i) {
    int oc = bx + ty + i * 8;          // output row = input col
    int orow = by + tx;                // output col = input row
    ot[(size_t)oc * rows + orow] = (bf16)t[tx][ty + i * 8];
  }
}

__global__ void noise_scale_kernel(const float* __restrict__ theta, float* __restrict__ ns, int n) {
  int i = blockIdx.x * blockDim.x + threadIdx.x;
  if (i < n) ns[i] = fabsf(sinf(2.0f * theta[i])) * 0.1f;
}

// ---------------- top-k (softmax + gumbel scores, bitonic sort), 1 block x 1024 thr ----------------
__global__ void select_topk_kernel(const float* __restrict__ vals, const float* __restrict__ gum,
                                   int k, int* __restrict__ out_idx, float* __restrict__ out_mask) {
  __shared__ float sv[NN];
  __shared__ unsigned long long keys[NN];
  __shared__ float red[1024];
  int tid = threadIdx.x;
  sv[tid] = vals[tid];
  sv[tid + 1024] = vals[tid + 1024];
  __syncthreads();
  red[tid] = fmaxf(sv[tid], sv[tid + 1024]);
  __syncthreads();
  for (int s = 512; s > 0; s >>= 1) {
    if (tid < s) red[tid] = fmaxf(red[tid], red[tid + s]);
    __syncthreads();
  }
  float vmax = red[0];
  __syncthreads();
  red[tid] = expf(sv[tid] - vmax) + expf(sv[tid + 1024] - vmax);
  __syncthreads();
  for (int s = 512; s > 0; s >>= 1) {
    if (tid < s) red[tid] += red[tid + s];
    __syncthreads();
  }
  float denom = red[0];
#pragma unroll
  for (int ii = 0; ii < 2; ++ii) {
    int i = tid + ii * 1024;
    float p  = expf(sv[i] - vmax) / denom;
    float sc = logf(p + 1e-8f) + gum[i];
    unsigned int b = __float_as_uint(sc);
    b = (b & 0x80000000u) ? ~b : (b | 0x80000000u);
    keys[i] = ((unsigned long long)b << 32) |
              (unsigned long long)(0xFFFFFFFFu - (unsigned)i);
  }
  __syncthreads();
  for (int kk = 2; kk <= NN; kk <<= 1) {
    for (int j = kk >> 1; j > 0; j >>= 1) {
#pragma unroll
      for (int ii = 0; ii < 2; ++ii) {
        int i = tid + ii * 1024;
        int l = i ^ j;
        if (l > i) {
          unsigned long long a = keys[i], c = keys[l];
          bool desc = ((i & kk) == 0);
          bool sw = desc ? (a < c) : (a > c);
          if (sw) { keys[i] = c; keys[l] = a; }
        }
      }
      __syncthreads();
    }
  }
  if (out_mask) { out_mask[tid] = 0.0f; out_mask[tid + 1024] = 0.0f; }
  __syncthreads();
#pragma unroll
  for (int ii = 0; ii < 2; ++ii) {
    int i = tid + ii * 1024;
    if (i < k) {
      int idx = (int)(0xFFFFFFFFu - (unsigned)(keys[i] & 0xFFFFFFFFull));
      out_idx[i] = idx;
      if (out_mask) out_mask[idx] = 1.0f;
    }
  }
}

// ---------------- init GEMM: st0 = (x @ Wi + bi) * input_mask ; s0 = tanh(st0) ----------------
__global__ __launch_bounds__(256)
void gemm_init_kernel(const float* __restrict__ x, const bf16* __restrict__ wiT,
                      const float* __restrict__ bi, const float* __restrict__ imask,
                      bf16* __restrict__ st, bf16* __restrict__ s) {
  __shared__ bf16 ldsA[BM * LDA];
  __shared__ bf16 ldsB[BN * LDB];
  int tid = threadIdx.x, lane = tid & 31, wid = tid >> 5;
  int bm = blockIdx.x * BM, bn = blockIdx.y * BN;
  int wm = (wid & 3) * 32, wn = (wid >> 2) * 32;
  f32x8 acc[2][2] = {};
  for (int kt = 0; kt < KIN; kt += BK) {
    stage_a_f32(ldsA, x, bm, kt, KIN, tid);
    stage_bt(ldsB, wiT, bn, kt, KIN, tid);
    __syncthreads();
    bf16x16 a0 = load_frag_a(ldsA, wm, lane);
    bf16x16 a1 = load_frag_a(ldsA, wm + 16, lane);
    bf16x16 b0 = load_frag_b(ldsB, wn, lane);
    bf16x16 b1 = load_frag_b(ldsB, wn + 16, lane);
    acc[0][0] = WMMA_BF16(a0, b0, acc[0][0]);
    acc[0][1] = WMMA_BF16(a0, b1, acc[0][1]);
    acc[1][0] = WMMA_BF16(a1, b0, acc[1][0]);
    acc[1][1] = WMMA_BF16(a1, b1, acc[1][1]);
    __syncthreads();
  }
  int hf = lane >> 4, nc = lane & 15;
#pragma unroll
  for (int ms = 0; ms < 2; ++ms)
#pragma unroll
    for (int ns = 0; ns < 2; ++ns)
#pragma unroll
      for (int r = 0; r < 8; ++r) {
        int row = bm + wm + ms * 16 + r + 8 * hf;
        int col = bn + wn + ns * 16 + nc;
        float v = (acc[ms][ns][r] + bi[col]) * imask[col];
        size_t o = (size_t)row * NN + col;
        st[o] = (bf16)v;
        s[o]  = (bf16)tanhf(v);
      }
}

// ---------------- fused step: st' = tanh(st@W + lam*((s@J)*s) + noise*ns) ----------------
__global__ __launch_bounds__(256)
void step_kernel(const bf16* __restrict__ stIn, const bf16* __restrict__ sIn,
                 const bf16* __restrict__ effwT, const bf16* __restrict__ jmT,
                 const float* __restrict__ noise, const float* __restrict__ nscale,
                 const float* __restrict__ lam_p,
                 bf16* __restrict__ stOut, bf16* __restrict__ sOut) {
  __shared__ bf16 ldsSt[BM * LDA];
  __shared__ bf16 ldsS [BM * LDA];
  __shared__ bf16 ldsW [BN * LDB];
  __shared__ bf16 ldsJ [BN * LDB];
  int tid = threadIdx.x, lane = tid & 31, wid = tid >> 5;
  int bm = blockIdx.x * BM, bn = blockIdx.y * BN;
  int wm = (wid & 3) * 32, wn = (wid >> 2) * 32;
  float lam = *lam_p;
#if HAVE_TDM
  unsigned ldsW_off = (unsigned)(uintptr_t)(void*)ldsW;
  unsigned ldsJ_off = (unsigned)(uintptr_t)(void*)ldsJ;
#endif
  f32x8 sig[2][2] = {};
  f32x8 ent[2][2] = {};
  for (int kt = 0; kt < NN; kt += BK) {
#if HAVE_TDM
    if (wid == 0) {  // wave 0 drives the Tensor Data Mover for both weight tiles
      tdm_load_tile_2d(ldsW_off, effwT + (size_t)bn * NN + kt, BK, BN, NN, NN, NN);
      tdm_load_tile_2d(ldsJ_off, jmT  + (size_t)bn * NN + kt, BK, BN, NN, NN, NN);
    }
#else
    stage_bt(ldsW, effwT, bn, kt, NN, tid);
    stage_bt(ldsJ, jmT,  bn, kt, NN, tid);
#endif
    if (kt + BK < NN) {  // gfx1250 global_prefetch_b8 of next state K-tile
      __builtin_prefetch(stIn + (size_t)(bm + (tid >> 1)) * NN + kt + BK + (tid & 1) * 16, 0, 1);
      __builtin_prefetch(sIn  + (size_t)(bm + (tid >> 1)) * NN + kt + BK + (tid & 1) * 16, 0, 1);
    }
    stage_a_bf16(ldsSt, stIn, bm, kt, NN, tid);
    stage_a_bf16(ldsS,  sIn,  bm, kt, NN, tid);
#if HAVE_TDM
    if (wid == 0) __builtin_amdgcn_s_wait_tensorcnt((short)0);
#endif
    __syncthreads();
    bf16x16 aSt0 = load_frag_a(ldsSt, wm,      lane);
    bf16x16 aSt1 = load_frag_a(ldsSt, wm + 16, lane);
    bf16x16 aS0  = load_frag_a(ldsS,  wm,      lane);
    bf16x16 aS1  = load_frag_a(ldsS,  wm + 16, lane);
    bf16x16 bW0  = load_frag_b(ldsW,  wn,      lane);
    bf16x16 bW1  = load_frag_b(ldsW,  wn + 16, lane);
    bf16x16 bJ0  = load_frag_b(ldsJ,  wn,      lane);
    bf16x16 bJ1  = load_frag_b(ldsJ,  wn + 16, lane);
    sig[0][0] = WMMA_BF16(aSt0, bW0, sig[0][0]);
    sig[0][1] = WMMA_BF16(aSt0, bW1, sig[0][1]);
    sig[1][0] = WMMA_BF16(aSt1, bW0, sig[1][0]);
    sig[1][1] = WMMA_BF16(aSt1, bW1, sig[1][1]);
    ent[0][0] = WMMA_BF16(aS0,  bJ0, ent[0][0]);
    ent[0][1] = WMMA_BF16(aS0,  bJ1, ent[0][1]);
    ent[1][0] = WMMA_BF16(aS1,  bJ0, ent[1][0]);
    ent[1][1] = WMMA_BF16(aS1,  bJ1, ent[1][1]);
    __syncthreads();
  }
  int hf = lane >> 4, nc = lane & 15;
#pragma unroll
  for (int ms = 0; ms < 2; ++ms)
#pragma unroll
    for (int ns = 0; ns < 2; ++ns)
#pragma unroll
      for (int r = 0; r < 8; ++r) {
        int row = bm + wm + ms * 16 + r + 8 * hf;
        int col = bn + wn + ns * 16 + nc;
        size_t o = (size_t)row * NN + col;
        float sv  = (float)sIn[o];
        float nz  = noise[o];
        float pre = sig[ms][ns][r] + lam * ent[ms][ns][r] * sv + nz * nscale[col];
        float sn  = tanhf(pre);
        stOut[o] = (bf16)sn;
        sOut[o]  = (bf16)tanhf(sn);
      }
}

// ---------------- output GEMM: out = st @ Wo + bo (f32, 512 cols) ----------------
__global__ __launch_bounds__(256)
void gemm_out_kernel(const bf16* __restrict__ st, const bf16* __restrict__ woT,
                     const float* __restrict__ bo, float* __restrict__ out) {
  __shared__ bf16 ldsA[BM * LDA];
  __shared__ bf16 ldsB[BN * LDB];
  int tid = threadIdx.x, lane = tid & 31, wid = tid >> 5;
  int bm = blockIdx.x * BM, bn = blockIdx.y * BN;
  int wm = (wid & 3) * 32, wn = (wid >> 2) * 32;
  f32x8 acc[2][2] = {};
  for (int kt = 0; kt < NN; kt += BK) {
    stage_a_bf16(ldsA, st, bm, kt, NN, tid);
    stage_bt(ldsB, woT, bn, kt, NN, tid);
    __syncthreads();
    bf16x16 a0 = load_frag_a(ldsA, wm, lane);
    bf16x16 a1 = load_frag_a(ldsA, wm + 16, lane);
    bf16x16 b0 = load_frag_b(ldsB, wn, lane);
    bf16x16 b1 = load_frag_b(ldsB, wn + 16, lane);
    acc[0][0] = WMMA_BF16(a0, b0, acc[0][0]);
    acc[0][1] = WMMA_BF16(a0, b1, acc[0][1]);
    acc[1][0] = WMMA_BF16(a1, b0, acc[1][0]);
    acc[1][1] = WMMA_BF16(a1, b1, acc[1][1]);
    __syncthreads();
  }
  int hf = lane >> 4, nc = lane & 15;
#pragma unroll
  for (int ms = 0; ms < 2; ++ms)
#pragma unroll
    for (int ns = 0; ns < 2; ++ns)
#pragma unroll
      for (int r = 0; r < 8; ++r) {
        int row = bm + wm + ms * 16 + r + 8 * hf;
        int col = bn + wn + ns * 16 + nc;
        out[(size_t)row * NOUT + col] = acc[ms][ns][r] + bo[col];
      }
}

// ---------------- column mean -> combined = tendency + |mean_b state| ----------------
__global__ void col_combined_kernel(const bf16* __restrict__ st, const float* __restrict__ tend,
                                    float* __restrict__ comb) {
  int j = blockIdx.x * blockDim.x + threadIdx.x;
  float s = 0.0f;
  for (int b = 0; b < BATCH; ++b) s += (float)st[(size_t)b * NN + j];
  comb[j] = tend[j] + fabsf(s * (1.0f / (float)BATCH));
}

// ---------------- host ----------------
extern "C" void kernel_launch(void* const* d_in, const int* in_sizes, int n_in,
                              void* d_out, int out_size, void* d_ws, size_t ws_size,
                              hipStream_t stream) {
  (void)in_sizes; (void)n_in; (void)out_size; (void)ws_size;
  const float* x      = (const float*)d_in[0];
  const float* theta  = (const float*)d_in[1];
  const float* sens   = (const float*)d_in[2];
  const float* otend  = (const float*)d_in[3];
  const float* cmask  = (const float*)d_in[4];
  const float* wts    = (const float*)d_in[5];
  const float* Jw     = (const float*)d_in[6];
  const float* lam    = (const float*)d_in[7];
  const float* Wi     = (const float*)d_in[8];
  const float* bi     = (const float*)d_in[9];
  const float* Wo     = (const float*)d_in[10];
  const float* bo     = (const float*)d_in[11];
  const float* g_in   = (const float*)d_in[12];
  const float* g_out  = (const float*)d_in[13];
  const float* snoise = (const float*)d_in[14];

  char* ws = (char*)d_ws;
  auto alloc = [&](size_t bytes) -> char* {
    char* p = ws;
    ws += (bytes + 255) & ~(size_t)255;
    return p;
  };
  bf16*  effwT  = (bf16*)alloc((size_t)NN * NN * 2);     // (weights*mask)^T  [n][k]
  bf16*  jmT    = (bf16*)alloc((size_t)NN * NN * 2);     // (J*mask)^T        [n][k]
  bf16*  wiT    = (bf16*)alloc((size_t)NN * KIN * 2);    // Wi^T [2048][1024]
  bf16*  woT    = (bf16*)alloc((size_t)NOUT * NN * 2);   // Wo^T [512][2048]
  bf16*  stA    = (bf16*)alloc((size_t)BATCH * NN * 2);
  bf16*  sA     = (bf16*)alloc((size_t)BATCH * NN * 2);
  bf16*  stB    = (bf16*)alloc((size_t)BATCH * NN * 2);
  bf16*  sB     = (bf16*)alloc((size_t)BATCH * NN * 2);
  float* nscale = (float*)alloc((size_t)NN * 4);
  float* imask  = (float*)alloc((size_t)NN * 4);
  float* comb   = (float*)alloc((size_t)NN * 4);

  float* outF   = (float*)d_out;
  int*   idxIn  = (int*)(outF + (size_t)BATCH * NOUT);
  int*   idxOut = idxIn + KIN;

  // prep: masked weights + all projections, transposed to [n][k] bf16
  dim3 gT(NN / 32, NN / 32);
  transpose_to_bf16_kernel<<<gT, 256, 0, stream>>>(wts, cmask, effwT, NN, NN);
  transpose_to_bf16_kernel<<<gT, 256, 0, stream>>>(Jw, cmask, jmT, NN, NN);
  transpose_to_bf16_kernel<<<dim3(NN / 32, KIN / 32), 256, 0, stream>>>(Wi, nullptr, wiT, KIN, NN);
  transpose_to_bf16_kernel<<<dim3(NOUT / 32, NN / 32), 256, 0, stream>>>(Wo, nullptr, woT, NN, NOUT);
  noise_scale_kernel<<<NN / 256, 256, 0, stream>>>(theta, nscale, NN);
  select_topk_kernel<<<1, 1024, 0, stream>>>(sens, g_in, KIN, idxIn, imask);

  // state0
  dim3 gFull(BATCH / BM, NN / BN);
  gemm_init_kernel<<<gFull, 256, 0, stream>>>(x, wiT, bi, imask, stA, sA);

  // 8 recurrent steps, double-buffered bf16 state
  bf16 *stI = stA, *sI = sA, *stO = stB, *sO = sB;
  for (int t = 0; t < TSTEP; ++t) {
    step_kernel<<<gFull, 256, 0, stream>>>(stI, sI, effwT, jmT,
                                           snoise + (size_t)t * BATCH * NN,
                                           nscale, lam, stO, sO);
    bf16* tmp;
    tmp = stI; stI = stO; stO = tmp;
    tmp = sI;  sI  = sO;  sO  = tmp;
  }

  // output selection + projection
  col_combined_kernel<<<NN / 256, 256, 0, stream>>>(stI, otend, comb);
  select_topk_kernel<<<1, 1024, 0, stream>>>(comb, g_out, NOUT, idxOut, nullptr);
  dim3 gOut(BATCH / BM, NOUT / BN);
  gemm_out_kernel<<<gOut, 256, 0, stream>>>(stI, woT, bo, outF);
}
